// GATClassifier_78357383348322
// MI455X (gfx1250) — matrix operand order repaired and checked by reading, not compiled, not run on hardware
//
#include <hip/hip_runtime.h>
#include <hip/hip_bf16.h>

typedef __attribute__((ext_vector_type(16))) _Float16 v16h;
typedef __attribute__((ext_vector_type(8)))  float    v8f;

// ---------------- helpers ----------------
__device__ __forceinline__ unsigned short f2h_bits(float f) {
    _Float16 h = (_Float16)f;
    union { _Float16 h; unsigned short s; } u; u.h = h; return u.s;
}
// order-preserving float -> uint key (for atomicMax on floats incl. negatives)
__device__ __forceinline__ unsigned fkey(float f) {
    unsigned u = __float_as_uint(f);
    return (u & 0x80000000u) ? ~u : (u | 0x80000000u);
}
__device__ __forceinline__ float fdecode(unsigned k) {
    unsigned u = (k & 0x80000000u) ? (k & 0x7fffffffu) : ~k;
    return __uint_as_float(u);
}

// ---------------- conversion kernels ----------------
__global__ void cvt_f32_to_f16(const float* __restrict__ in,
                               unsigned short* __restrict__ out, int n) {
    int i = blockIdx.x * blockDim.x + threadIdx.x;
    if (i < n) out[i] = f2h_bits(in[i]);
}

// W [K x M] row-major f32  ->  Wt [M x K] row-major f16  (so B loads are contiguous)
__global__ void cvt_w_transpose(const float* __restrict__ W,
                                unsigned short* __restrict__ Wt, int K, int M) {
    int t = blockIdx.x * blockDim.x + threadIdx.x;
    if (t >= K * M) return;
    int i = t / M, j = t - i * M;
    Wt[(size_t)j * K + i] = f2h_bits(W[(size_t)i * M + j]);
}

// ---------------- WMMA GEMM: H[N,M] = X[N,K](f16) * Wt[M,K]^T(f16), f32 accum ----------------
// one wave per 16x64 output strip (4 accumulators share one A tile per K-step);
// K multiple of 32; M multiple of 64
#define CT 4
__global__ void wmma_gemm_f16(const unsigned short* __restrict__ X,
                              const unsigned short* __restrict__ Wt,
                              float* __restrict__ Hout,
                              int Nrows, int K, int M) {
    int wave = blockIdx.x * (blockDim.x >> 5) + (threadIdx.x >> 5);
    int lane = threadIdx.x & 31;
    int colGroups = M / (16 * CT);
    int rowTiles  = (Nrows + 15) >> 4;
    if (wave >= rowTiles * colGroups) return;
    int rowTile = wave / colGroups;
    int colGrp  = wave - rowTile * colGroups;
    int row0 = rowTile << 4;
    int col0 = colGrp * (16 * CT);
    int m16  = lane & 15;
    int half = lane >> 4;                 // 0 or 1

    int arow_idx = row0 + m16;
    if (arow_idx >= Nrows) arow_idx = Nrows - 1;   // clamp (N=50000 is exact anyway)

    v8f c[CT];
    #pragma unroll
    for (int t = 0; t < CT; ++t) c[t] = (v8f){};

    for (int kk = 0; kk < K; kk += 32) {
        // A: 16x32 f16 tile, lane (m16, half). ISA layout:
        //   e=2j,2j+1 -> K = kk + (j>=4?16:0) + half*8 + 2*(j&3) (+0/+1)
        union { v16h v; unsigned int u[8]; } a;
        const unsigned short* arow = X + (size_t)arow_idx * K + kk + (half << 3);
        #pragma unroll
        for (int j = 0; j < 8; ++j) {
            int kbase = ((j & 4) << 2) + ((j & 3) << 1);
            a.u[j] = *(const unsigned int*)(arow + kbase);
        }
        // B: 32x16 f16 tiles, lane = column n=m16; K = kk + (e>=8?16:0) + half*8 + (e&7) (same as A)
        union { v16h v; uint4 q[2]; } b[CT];
        #pragma unroll
        for (int t = 0; t < CT; ++t) {
            const uint4* bq = (const uint4*)(Wt + (size_t)(col0 + (t << 4) + m16) * K
                                             + kk + (half << 3));
            b[t].q[0] = bq[0];
            b[t].q[1] = bq[2];
        }
        #pragma unroll
        for (int t = 0; t < CT; ++t)
            c[t] = __builtin_amdgcn_wmma_f32_16x16x32_f16(false, a.v, false, b[t].v,
                                                          (short)0, c[t], false, false);
    }
    // C/D layout: VGPR v -> M = v + half*8, N = m16
    #pragma unroll
    for (int t = 0; t < CT; ++t) {
        #pragma unroll
        for (int v = 0; v < 8; ++v) {
            int mo = row0 + v + (half << 3);
            if (mo < Nrows) Hout[(size_t)mo * M + col0 + (t << 4) + m16] = c[t][v];
        }
    }
}

// ---------------- attention logits per node/head ----------------
__global__ void alpha_kernel(const float* __restrict__ Hg,       // [N, H*D]
                             const float* __restrict__ a_src,    // [H, D]
                             const float* __restrict__ a_dst,    // [H, D]
                             float* __restrict__ as_, float* __restrict__ ad_,
                             int N, int H, int D) {
    int t = blockIdx.x * blockDim.x + threadIdx.x;
    if (t >= N * H) return;
    int n = t / H, h = t - n * H;
    const float* hv = Hg + (size_t)n * H * D + h * D;
    const float* vs = a_src + h * D;
    const float* vd = a_dst + h * D;
    float s = 0.f, d = 0.f;
    for (int i = 0; i < D; ++i) { float x = hv[i]; s += x * vs[i]; d += x * vd[i]; }
    as_[t] = s; ad_[t] = d;
}

__device__ __forceinline__ float edge_logit(const int* __restrict__ ei,
                                            const float* __restrict__ as_,
                                            const float* __restrict__ ad_,
                                            int e, int E, int H, int h,
                                            int& s_out, int& d_out) {
    int s = (e < E) ? ei[e]     : (e - E);   // self-loop tail
    int d = (e < E) ? ei[E + e] : (e - E);
    s_out = s; d_out = d;
    float v = as_[s * H + h] + ad_[d * H + h];
    return v > 0.f ? v : 0.2f * v;           // leaky_relu(0.2)
}

__global__ void edge_max_kernel(const int* __restrict__ ei,
                                const float* __restrict__ as_, const float* __restrict__ ad_,
                                unsigned* __restrict__ mb, int E, int N, int H) {
    int t = blockIdx.x * blockDim.x + threadIdx.x;
    if (t >= (E + N) * H) return;
    int e = t / H, h = t - e * H, s, d;
    float v = edge_logit(ei, as_, ad_, e, E, H, h, s, d);
    atomicMax(&mb[d * H + h], fkey(v));
}

__global__ void edge_sum_kernel(const int* __restrict__ ei,
                                const float* __restrict__ as_, const float* __restrict__ ad_,
                                const unsigned* __restrict__ mb, float* __restrict__ den,
                                int E, int N, int H) {
    int t = blockIdx.x * blockDim.x + threadIdx.x;
    if (t >= (E + N) * H) return;
    int e = t / H, h = t - e * H, s, d;
    float v = edge_logit(ei, as_, ad_, e, E, H, h, s, d);
    float m = fdecode(mb[d * H + h]);
    atomicAdd(&den[d * H + h], __expf(v - m));
}

__global__ void edge_aggregate_kernel(const int* __restrict__ ei,
                                      const float* __restrict__ as_, const float* __restrict__ ad_,
                                      const unsigned* __restrict__ mb, const float* __restrict__ den,
                                      const float* __restrict__ Hg, float* __restrict__ agg,
                                      int E, int N, int H, int D) {
    int t = blockIdx.x * blockDim.x + threadIdx.x;
    if (t >= (E + N) * H) return;
    int e = t / H, h = t - e * H, s, d;
    float v = edge_logit(ei, as_, ad_, e, E, H, h, s, d);
    float m = fdecode(mb[d * H + h]);
    float w = __expf(v - m) / (den[d * H + h] + 1e-16f);
    const float* hrow = Hg  + (size_t)s * H * D + h * D;
    float*       orow = agg + (size_t)d * H * D + h * D;
    for (int i = 0; i < D; i += 4) {
        float4 x4 = *(const float4*)(hrow + i);
        atomicAdd(orow + i + 0, w * x4.x);
        atomicAdd(orow + i + 1, w * x4.y);
        atomicAdd(orow + i + 2, w * x4.z);
        atomicAdd(orow + i + 3, w * x4.w);
    }
}

// head-mean + bias (+ELU)
__global__ void finalize_kernel(const float* __restrict__ agg, const float* __restrict__ bias,
                                float* __restrict__ feat, int N, int H, int D, int applyElu) {
    int t = blockIdx.x * blockDim.x + threadIdx.x;
    if (t >= N * D) return;
    int n = t / D, d = t - n * D;
    float s = 0.f;
    for (int h = 0; h < H; ++h) s += agg[(size_t)n * H * D + h * D + d];
    s = s / (float)H + bias[d];
    if (applyElu) s = (s > 0.f) ? s : (__expf(s) - 1.f);
    feat[t] = s;
}

__global__ void pool_kernel(const float* __restrict__ feat, const int* __restrict__ batch,
                            float* __restrict__ pooled, float* __restrict__ counts,
                            int N, int D) {
    int t = blockIdx.x * blockDim.x + threadIdx.x;
    if (t >= N * D) return;
    int n = t / D, d = t - n * D;
    int b = batch[n];
    atomicAdd(&pooled[b * D + d], feat[t]);
    if (d == 0) atomicAdd(&counts[b], 1.f);
}

__global__ void mlp_kernel(const float* __restrict__ pooled, const float* __restrict__ counts,
                           const float* __restrict__ cW1, const float* __restrict__ cb1,
                           const float* __restrict__ cW2, const float* __restrict__ cb2,
                           float* __restrict__ out, int G, int D, int C1, int C2) {
    int g = blockIdx.x * blockDim.x + threadIdx.x;
    if (g >= G) return;
    float inv = 1.f / fmaxf(counts[g], 1.f);
    for (int k = 0; k < C2; ++k) out[g * C2 + k] = cb2[k];
    for (int j = 0; j < C1; ++j) {
        float z = cb1[j];
        for (int d = 0; d < D; ++d) z += pooled[g * D + d] * inv * cW1[d * C1 + j];
        z = fmaxf(z, 0.f);
        for (int k = 0; k < C2; ++k) out[g * C2 + k] += z * cW2[j * C2 + k];
    }
}

// ---------------- driver ----------------
extern "C" void kernel_launch(void* const* d_in, const int* in_sizes, int n_in,
                              void* d_out, int out_size, void* d_ws, size_t ws_size,
                              hipStream_t stream) {
    const float* x      = (const float*)d_in[0];
    const int*   ei     = (const int*)  d_in[1];
    const int*   batch  = (const int*)  d_in[2];
    const float* W0     = (const float*)d_in[3];
    const float* a_src0 = (const float*)d_in[4];
    const float* a_dst0 = (const float*)d_in[5];
    const float* b0     = (const float*)d_in[6];
    const float* W1     = (const float*)d_in[7];
    const float* a_src1 = (const float*)d_in[8];
    const float* a_dst1 = (const float*)d_in[9];
    const float* b1     = (const float*)d_in[10];
    const float* W2     = (const float*)d_in[11];
    const float* a_src2 = (const float*)d_in[12];
    const float* a_dst2 = (const float*)d_in[13];
    const float* b2     = (const float*)d_in[14];
    const float* cW1    = (const float*)d_in[15];
    const float* cb1    = (const float*)d_in[16];
    const float* cW2    = (const float*)d_in[17];
    const float* cb2    = (const float*)d_in[18];

    const int N   = in_sizes[2];
    const int E   = in_sizes[1] / 2;
    const int FIN = in_sizes[0] / N;        // 128
    const int D   = in_sizes[6];            // 64
    const int H0  = in_sizes[4] / D;        // 4
    const int H1  = in_sizes[8] / D;        // 4
    const int H2  = in_sizes[12] / D;       // 1
    const int G   = 64;                     // NUM_GRAPHS
    const int C1  = in_sizes[16];           // 32
    const int C2  = in_sizes[18];           // 2
    const int Hmax = (H0 > H1) ? H0 : H1;
    const int MoutMax = Hmax * D;           // 256

    // carve workspace
    char* base = (char*)d_ws; size_t off = 0;
    auto carve = [&](size_t bytes) -> char* {
        char* p = base + off; off += (bytes + 255) & ~(size_t)255; return p;
    };
    float*          hgemm  = (float*)         carve((size_t)N * MoutMax * 4);  // GEMM output [N, H*D]
    float*          agg    = (float*)         carve((size_t)N * MoutMax * 4);  // scatter accumulator
    float*          feat   = (float*)         carve((size_t)N * D * 4);        // layer output [N, D]
    unsigned short* xin    = (unsigned short*)carve((size_t)N * FIN * 2);      // f16 GEMM input
    unsigned short* wt     = (unsigned short*)carve((size_t)FIN * MoutMax * 2);
    float*          as_    = (float*)         carve((size_t)N * Hmax * 4);
    float*          ad_    = (float*)         carve((size_t)N * Hmax * 4);
    unsigned*       mb     = (unsigned*)      carve((size_t)N * Hmax * 4);
    float*          den    = (float*)         carve((size_t)N * Hmax * 4);
    float*          pooled = (float*)         carve((size_t)G * D * 4);
    float*          counts = (float*)         carve((size_t)G * 4);

    const int TB = 256;
    auto blocks = [&](long long n) -> unsigned { return (unsigned)((n + TB - 1) / TB); };

    auto run_layer = [&](const unsigned short* Xf16, int K, const float* W,
                         const float* a_s, const float* a_d, const float* bias,
                         int H, int applyElu) {
        int Mout = H * D;
        cvt_w_transpose<<<blocks((long long)K * Mout), TB, 0, stream>>>(W, wt, K, Mout);
        int rowTiles = (N + 15) / 16, colGroups = Mout / (16 * CT);
        long long waves = (long long)rowTiles * colGroups;
        wmma_gemm_f16<<<(unsigned)((waves + 7) / 8), TB, 0, stream>>>(Xf16, wt, hgemm, N, K, Mout);
        alpha_kernel<<<blocks((long long)N * H), TB, 0, stream>>>(hgemm, a_s, a_d, as_, ad_, N, H, D);
        hipMemsetAsync(mb,  0, (size_t)N * H * 4, stream);
        hipMemsetAsync(den, 0, (size_t)N * H * 4, stream);
        long long ET = (long long)(E + N) * H;
        edge_max_kernel<<<blocks(ET), TB, 0, stream>>>(ei, as_, ad_, mb, E, N, H);
        edge_sum_kernel<<<blocks(ET), TB, 0, stream>>>(ei, as_, ad_, mb, den, E, N, H);
        hipMemsetAsync(agg, 0, (size_t)N * Mout * 4, stream);
        edge_aggregate_kernel<<<blocks(ET), TB, 0, stream>>>(ei, as_, ad_, mb, den, hgemm, agg, E, N, H, D);
        finalize_kernel<<<blocks((long long)N * D), TB, 0, stream>>>(agg, bias, feat, N, H, D, applyElu);
    };

    // layer 0
    cvt_f32_to_f16<<<blocks((long long)N * FIN), TB, 0, stream>>>(x, xin, N * FIN);
    run_layer(xin, FIN, W0, a_src0, a_dst0, b0, H0, 1);
    // layer 1
    cvt_f32_to_f16<<<blocks((long long)N * D), TB, 0, stream>>>(feat, xin, N * D);
    run_layer(xin, D, W1, a_src1, a_dst1, b1, H1, 1);
    // layer 2 (single head, no ELU)
    cvt_f32_to_f16<<<blocks((long long)N * D), TB, 0, stream>>>(feat, xin, N * D);
    run_layer(xin, D, W2, a_src2, a_dst2, b2, H2, 0);

    // global mean pool + classifier
    hipMemsetAsync(pooled, 0, (size_t)G * D * 4, stream);
    hipMemsetAsync(counts, 0, (size_t)G * 4, stream);
    pool_kernel<<<blocks((long long)N * D), TB, 0, stream>>>(feat, batch, pooled, counts, N, D);
    mlp_kernel<<<1, G, 0, stream>>>(pooled, counts, cW1, cb1, cW2, cb2, (float*)d_out, G, D, C1, C2);
    (void)n_in; (void)out_size; (void)ws_size;
}
